// MultiHeadAttentionV2_61125974556867
// MI455X (gfx1250) — matrix operand order repaired and checked by reading, not compiled, not run on hardware
//
#include <hip/hip_runtime.h>

typedef __bf16 bf16_t;
typedef __attribute__((ext_vector_type(16))) __bf16 v16bf;
typedef __attribute__((ext_vector_type(4)))  __bf16 v4bf;
typedef __attribute__((ext_vector_type(2)))  __bf16 v2bf;
typedef __attribute__((ext_vector_type(8)))  float  v8f;
typedef __attribute__((ext_vector_type(4)))  float  v4f;
typedef __attribute__((ext_vector_type(2)))  float  v2f;

namespace {
constexpr int   kB    = 4;
constexpr int   kT    = 2048;
constexpr int   kDim  = 1024;
constexpr int   kH    = 16;
constexpr int   kDh   = 64;
constexpr int   kN3   = 3 * kDim;
constexpr float kScale = 0.125f;   // HEAD_DIM^-0.5 = 64^-0.5
}

// ---------------------------------------------------------------------------
// WMMA helpers (CDNA5 wave32, v_wmma_f32_16x16x32_bf16)
// ---------------------------------------------------------------------------
__device__ __forceinline__ v8f wmma_bf16(v16bf a, v16bf b, v8f c) {
  return __builtin_amdgcn_wmma_f32_16x16x32_bf16(false, a, false, b,
                                                 (short)0, c, false, false);
}

// A-operand (16x32 bf16): lane half h -> K chunks {8h..8h+7, 16+8h..16+8h+7}
__device__ __forceinline__ v16bf frag_a(const bf16_t* __restrict__ row, int h) {
  union { v16bf v; uint2 u[4]; } f;
  f.u[0] = *(const uint2*)(row + 8 * h);
  f.u[1] = *(const uint2*)(row + 8 * h + 4);
  f.u[2] = *(const uint2*)(row + 16 + 8 * h);
  f.u[3] = *(const uint2*)(row + 16 + 8 * h + 4);
  return f.v;
}

// B-operand (32x16 bf16): lane half h -> K chunks {8h..8h+7, 16+8h..16+8h+7}
__device__ __forceinline__ v16bf frag_b(const bf16_t* __restrict__ row, int h) {
  union { v16bf v; uint2 u[4]; } f;
  const bf16_t* p = row + 8 * h;
  f.u[0] = *(const uint2*)(p);
  f.u[1] = *(const uint2*)(p + 4);
  f.u[2] = *(const uint2*)(p + 16);
  f.u[3] = *(const uint2*)(p + 20);
  return f.v;
}

// Packed fp32 -> bf16 converts: __builtin_convertvector lowers to
// v_cvt_pk_bf16_f32 pairs (the union/byte-wise form defeated the matcher).
__device__ __forceinline__ uint2 pack4(float4 f) {
  v4f vf = {f.x, f.y, f.z, f.w};
  union { v4bf v; uint2 u; } x;
  x.v = __builtin_convertvector(vf, v4bf);
  return x.u;
}

__device__ __forceinline__ unsigned pack2(float a, float b) {
  v2f vf = {a, b};
  union { v2bf v; unsigned u; } x;
  x.v = __builtin_convertvector(vf, v2bf);
  return x.u;
}

// ---------------------------------------------------------------------------
// Kernel 1: QKV projection  out = x @ qkv_w^T + qkv_b  (fp32 in, bf16 out)
//   Q -> [B,H,T,64] (pre-scaled by 1/sqrt(64)), K -> [B,H,T,64], V -> [B,H,64,T]
// Block = 256 threads (8 waves), tile 128(M) x 128(N), K-step 64 (2 chunks).
// Wave tile 32x64: acc[2][4], 16 WMMAs per barrier pair.
// ---------------------------------------------------------------------------
__global__ __launch_bounds__(256) void mha_qkv_kernel(
    const float* __restrict__ x, const float* __restrict__ w,
    const float* __restrict__ bias,
    bf16_t* __restrict__ qws, bf16_t* __restrict__ kws,
    bf16_t* __restrict__ vtws) {
  __shared__ bf16_t As[128][68];   // 128 x 64 (+4 pad)
  __shared__ bf16_t Bs[128][68];

  const int tid  = threadIdx.x;
  const int lane = tid & 31, wave = tid >> 5;
  const int h = lane >> 4, ln = lane & 15;
  const int n0 = blockIdx.x * 128, m0 = blockIdx.y * 128;
  const int wm = (wave >> 1) * 32, wn = (wave & 1) * 64;

  // Strength-reduced staging addressing: row = r0 + 16*i, fixed column.
  const int r0 = tid >> 4, c4 = (tid & 15) * 4;
  const float* px = x + (size_t)(m0 + r0) * kDim + c4;
  const float* pw = w + (size_t)(n0 + r0) * kDim + c4;
  bf16_t* sa = &As[r0][c4];
  bf16_t* sb = &Bs[r0][c4];

  v8f acc[2][4] = {};

  for (int kk = 0; kk < kDim; kk += 64) {
    __syncthreads();
#pragma unroll
    for (int i = 0; i < 8; ++i) {           // 2048 4-elem chunks / 256 threads
      const size_t go = (size_t)i * 16 * kDim;
      const int    so = i * 16 * 68;
      *(uint2*)(sa + so) = pack4(*(const float4*)(px + go));
      *(uint2*)(sb + so) = pack4(*(const float4*)(pw + go));
    }
    px += 64; pw += 64;
    __syncthreads();
#pragma unroll
    for (int kc = 0; kc < 2; ++kc) {
      const v16bf a0 = frag_a(&As[wm + ln][32 * kc], h);
      const v16bf a1 = frag_a(&As[wm + 16 + ln][32 * kc], h);
      const v16bf b0 = frag_b(&Bs[wn + ln][32 * kc], h);
      const v16bf b1 = frag_b(&Bs[wn + 16 + ln][32 * kc], h);
      const v16bf b2 = frag_b(&Bs[wn + 32 + ln][32 * kc], h);
      const v16bf b3 = frag_b(&Bs[wn + 48 + ln][32 * kc], h);
      acc[0][0] = wmma_bf16(a0, b0, acc[0][0]);
      acc[1][0] = wmma_bf16(a1, b0, acc[1][0]);
      acc[0][1] = wmma_bf16(a0, b1, acc[0][1]);
      acc[1][1] = wmma_bf16(a1, b1, acc[1][1]);
      acc[0][2] = wmma_bf16(a0, b2, acc[0][2]);
      acc[1][2] = wmma_bf16(a1, b2, acc[1][2]);
      acc[0][3] = wmma_bf16(a0, b3, acc[0][3]);
      acc[1][3] = wmma_bf16(a1, b3, acc[1][3]);
    }
  }

  // ---- Epilogue ----
  // region (q/k/v) uniform per block (128 | 1024); batch index uniform per
  // wave (32-row wave tile never straddles a T=2048 boundary); head uniform
  // per 16-wide j-tile. Hoist everything, branch exactly once.
  const int region = n0 >> 10;
  const int bidx  = (m0 + wm) >> 11;
  const int tbase = (m0 + wm) & (kT - 1);
  float bv[4];
  int   headj[4], dj[4];
#pragma unroll
  for (int j = 0; j < 4; ++j) {
    const int n = n0 + wn + j * 16 + ln;
    bv[j]    = bias[n];
    headj[j] = (n & 1023) >> 6;
    dj[j]    = n & 63;
  }

  if (region == 0) {                 // Q: [B,H,T,64], pre-scaled
#pragma unroll
    for (int j = 0; j < 4; ++j) {
      bf16_t* base = qws + (size_t)(bidx * kH + headj[j]) * kT * kDh + dj[j];
#pragma unroll
      for (int i = 0; i < 2; ++i) {
        const int t0 = tbase + i * 16 + 8 * h;
#pragma unroll
        for (int r = 0; r < 8; ++r)
          base[(unsigned)(t0 + r) * kDh] = (bf16_t)((acc[i][j][r] + bv[j]) * kScale);
      }
    }
  } else if (region == 1) {          // K: [B,H,T,64]
#pragma unroll
    for (int j = 0; j < 4; ++j) {
      bf16_t* base = kws + (size_t)(bidx * kH + headj[j]) * kT * kDh + dj[j];
#pragma unroll
      for (int i = 0; i < 2; ++i) {
        const int t0 = tbase + i * 16 + 8 * h;
#pragma unroll
        for (int r = 0; r < 8; ++r)
          base[(unsigned)(t0 + r) * kDh] = (bf16_t)(acc[i][j][r] + bv[j]);
      }
    }
  } else {                           // V^T: [B,H,64,T]; rows pack in t pairs
#pragma unroll
    for (int j = 0; j < 4; ++j) {
      bf16_t* base =
          vtws + ((size_t)(bidx * kH + headj[j]) * kDh + dj[j]) * kT;
#pragma unroll
      for (int i = 0; i < 2; ++i) {
        const int t0 = tbase + i * 16 + 8 * h;   // even -> b32 aligned
#pragma unroll
        for (int r = 0; r < 4; ++r)
          *(unsigned*)(base + t0 + 2 * r) =
              pack2(acc[i][j][2 * r] + bv[j], acc[i][j][2 * r + 1] + bv[j]);
      }
    }
  }
}

// ---------------------------------------------------------------------------
// Kernel 2: flash attention.  grid = (B*H, T/128), block = 256 (8 waves).
// Each wave: 16 q rows; online softmax over key blocks of 64.
// Per key block per wave: 8 WMMAs (S) + 8 WMMAs (AV).
// ---------------------------------------------------------------------------
__global__ __launch_bounds__(256) void mha_attn_kernel(
    const bf16_t* __restrict__ q, const bf16_t* __restrict__ k,
    const bf16_t* __restrict__ vt, const int* __restrict__ mask,
    bf16_t* __restrict__ aout) {
  __shared__ bf16_t Ks[64][68];       // 64 keys x 64 d    (+4 pad)
  __shared__ bf16_t Vs[64][68];       // 64 d    x 64 keys (+4 pad)
  __shared__ bf16_t Ps[8][16][68];    // per-wave P tile 16 x 64

  const int tid  = threadIdx.x;
  const int lane = tid & 31, wave = tid >> 5;
  const int h = lane >> 4, ln = lane & 15;
  const int bh = blockIdx.x;
  const int b = bh >> 4, head = bh & 15;
  const int q0 = blockIdx.y * 128 + wave * 16;

  // Persistent Q A-fragments (d = 0..31 and 32..63), loaded straight from HBM
  const bf16_t* qrow = q + ((size_t)bh * kT + q0 + ln) * kDh;
  const v16bf qa0 = frag_a(qrow, h);
  const v16bf qa1 = frag_a(qrow + 32, h);

  v8f o[4] = {};
  float mrow[8], lrow[8];
#pragma unroll
  for (int r = 0; r < 8; ++r) { mrow[r] = -__builtin_inff(); lrow[r] = 0.f; }
  const int* mp = mask + (size_t)b * kT;

  // Strength-reduced staging pointers (row = r0 + 16*i, fixed column)
  const int r0 = tid >> 4, c4 = (tid & 15) * 4;
  const bf16_t* pk = k  + ((size_t)bh * kT + r0) * kDh + c4;   // +64*kDh per blk
  const bf16_t* pv = vt + ((size_t)bh * kDh + r0) * kT + c4;   // +64 per blk
  bf16_t* sk = &Ks[r0][c4];
  bf16_t* sv = &Vs[r0][c4];

  for (int kb = 0; kb < kT; kb += 64) {
    __syncthreads();                      // prior AV reads of Ks/Vs done
#pragma unroll
    for (int i = 0; i < 4; ++i) {         // stage K (64x64) and V^T (64x64)
      *(uint2*)(sk + i * 16 * 68) = *(const uint2*)(pk + (size_t)i * 16 * kDh);
      *(uint2*)(sv + i * 16 * 68) = *(const uint2*)(pv + (size_t)i * 16 * kT);
    }
    pk += 64 * kDh; pv += 64;
    if (kb + 64 < kT) {                   // hint next key block toward caches
      __builtin_prefetch(pk + (size_t)(tid >> 2) * kDh, 0, 1);
      __builtin_prefetch(pv + (size_t)(tid >> 2) * kT, 0, 1);
    }
    __syncthreads();                      // staging visible

    // S = Q K^T : four 16x16 key tiles, contraction over d = 64 (2 WMMAs each)
    v8f s[4] = {};
#pragma unroll
    for (int tj = 0; tj < 4; ++tj) {
      v16bf kb0 = frag_b(&Ks[tj * 16 + ln][0], h);    // d 0..31
      v16bf kb1 = frag_b(&Ks[tj * 16 + ln][32], h);   // d 32..63
      s[tj] = wmma_bf16(qa0, kb0, s[tj]);
      s[tj] = wmma_bf16(qa1, kb1, s[tj]);
    }

    // Key mask (column depends only on lane) -> branchless select
    int mv[4];
#pragma unroll
    for (int tj = 0; tj < 4; ++tj) mv[tj] = mp[kb + tj * 16 + ln];
#pragma unroll
    for (int tj = 0; tj < 4; ++tj)
#pragma unroll
      for (int r = 0; r < 8; ++r)
        s[tj][r] = (mv[tj] != 0) ? s[tj][r] : -__builtin_inff();

    // Online softmax; each row lives in 16 lanes of one half -> xor 1,2,4,8
#pragma unroll
    for (int r = 0; r < 8; ++r) {
      float smax = fmaxf(fmaxf(s[0][r], s[1][r]), fmaxf(s[2][r], s[3][r]));
#pragma unroll
      for (int off = 8; off >= 1; off >>= 1)
        smax = fmaxf(smax, __shfl_xor(smax, off, 32));
      const float mnew = fmaxf(mrow[r], smax);
      float p[4], rs = 0.f;
#pragma unroll
      for (int tj = 0; tj < 4; ++tj) { p[tj] = __expf(s[tj][r] - mnew); rs += p[tj]; }
#pragma unroll
      for (int off = 8; off >= 1; off >>= 1)
        rs += __shfl_xor(rs, off, 32);
      const float alpha = __expf(mrow[r] - mnew);
      lrow[r] = lrow[r] * alpha + rs;
      mrow[r] = mnew;
#pragma unroll
      for (int j = 0; j < 4; ++j) o[j][r] *= alpha;
      // C-layout -> LDS P tile (row-major 16x64) for A-operand reload
      bf16_t* prow = &Ps[wave][r + 8 * h][ln];
#pragma unroll
      for (int tj = 0; tj < 4; ++tj) prow[tj * 16] = (bf16_t)p[tj];
    }
    __syncthreads();                      // cross-lane Ps visibility

    // O += P @ V : contraction over 64 keys (2 chunks), 4 output d-tiles
    const v16bf pa0 = frag_a(&Ps[wave][ln][0], h);    // keys 0..31
    const v16bf pa1 = frag_a(&Ps[wave][ln][32], h);   // keys 32..63
#pragma unroll
    for (int j = 0; j < 4; ++j) {
      v16bf vb0 = frag_b(&Vs[j * 16 + ln][0], h);
      v16bf vb1 = frag_b(&Vs[j * 16 + ln][32], h);
      o[j] = wmma_bf16(pa0, vb0, o[j]);
      o[j] = wmma_bf16(pa1, vb1, o[j]);
    }
  }

  // Normalize and write attn output as bf16 [B, T, H*64]
#pragma unroll
  for (int r = 0; r < 8; ++r) {
    const float inv = 1.0f / lrow[r];
    const int t = q0 + r + 8 * h;
    bf16_t* orow = aout + ((size_t)b * kT + t) * kDim + head * kDh + ln;
#pragma unroll
    for (int j = 0; j < 4; ++j)
      orow[j * 16] = (bf16_t)(o[j][r] * inv);
  }
}

// ---------------------------------------------------------------------------
// Kernel 3: output projection  out = attn_bf16 @ proj_w^T + proj_b (fp32 out)
// Same 128x128 / k-step-64 tiling as kernel 1.
// ---------------------------------------------------------------------------
__global__ __launch_bounds__(256) void mha_proj_kernel(
    const bf16_t* __restrict__ a, const float* __restrict__ w,
    const float* __restrict__ bias, float* __restrict__ out) {
  __shared__ bf16_t As[128][68];
  __shared__ bf16_t Bs[128][68];

  const int tid  = threadIdx.x;
  const int lane = tid & 31, wave = tid >> 5;
  const int h = lane >> 4, ln = lane & 15;
  const int n0 = blockIdx.x * 128, m0 = blockIdx.y * 128;
  const int wm = (wave >> 1) * 32, wn = (wave & 1) * 64;

  const int r0 = tid >> 4, c4 = (tid & 15) * 4;
  const bf16_t* pa = a + (size_t)(m0 + r0) * kDim + c4;
  const float*  pw = w + (size_t)(n0 + r0) * kDim + c4;
  bf16_t* sa = &As[r0][c4];
  bf16_t* sb = &Bs[r0][c4];

  v8f acc[2][4] = {};

  for (int kk = 0; kk < kDim; kk += 64) {
    __syncthreads();
#pragma unroll
    for (int i = 0; i < 8; ++i) {
      const size_t go = (size_t)i * 16 * kDim;
      const int    so = i * 16 * 68;
      *(uint2*)(sa + so) = *(const uint2*)(pa + go);      // already bf16
      *(uint2*)(sb + so) = pack4(*(const float4*)(pw + go));
    }
    pa += 64; pw += 64;
    __syncthreads();
#pragma unroll
    for (int kc = 0; kc < 2; ++kc) {
      const v16bf a0 = frag_a(&As[wm + ln][32 * kc], h);
      const v16bf a1 = frag_a(&As[wm + 16 + ln][32 * kc], h);
      const v16bf b0 = frag_b(&Bs[wn + ln][32 * kc], h);
      const v16bf b1 = frag_b(&Bs[wn + 16 + ln][32 * kc], h);
      const v16bf b2 = frag_b(&Bs[wn + 32 + ln][32 * kc], h);
      const v16bf b3 = frag_b(&Bs[wn + 48 + ln][32 * kc], h);
      acc[0][0] = wmma_bf16(a0, b0, acc[0][0]);
      acc[1][0] = wmma_bf16(a1, b0, acc[1][0]);
      acc[0][1] = wmma_bf16(a0, b1, acc[0][1]);
      acc[1][1] = wmma_bf16(a1, b1, acc[1][1]);
      acc[0][2] = wmma_bf16(a0, b2, acc[0][2]);
      acc[1][2] = wmma_bf16(a1, b2, acc[1][2]);
      acc[0][3] = wmma_bf16(a0, b3, acc[0][3]);
      acc[1][3] = wmma_bf16(a1, b3, acc[1][3]);
    }
  }

  // Straight-line epilogue: fp32 stores, bias hoisted
  float bv[4];
#pragma unroll
  for (int j = 0; j < 4; ++j) bv[j] = bias[n0 + wn + j * 16 + ln];
#pragma unroll
  for (int i = 0; i < 2; ++i) {
#pragma unroll
    for (int r = 0; r < 8; ++r) {
      const int m = m0 + wm + i * 16 + r + 8 * h;
      float* orow = out + (size_t)m * kDim + n0 + wn + ln;
#pragma unroll
      for (int j = 0; j < 4; ++j)
        orow[j * 16] = acc[i][j][r] + bv[j];
    }
  }
}

// ---------------------------------------------------------------------------
// Host launcher
// ---------------------------------------------------------------------------
extern "C" void kernel_launch(void* const* d_in, const int* in_sizes, int n_in,
                              void* d_out, int out_size, void* d_ws,
                              size_t ws_size, hipStream_t stream) {
  (void)in_sizes; (void)n_in; (void)out_size; (void)ws_size;
  const float* x      = (const float*)d_in[0];
  const int*   mask   = (const int*)d_in[1];
  const float* qkv_w  = (const float*)d_in[2];
  const float* qkv_b  = (const float*)d_in[3];
  const float* proj_w = (const float*)d_in[4];
  const float* proj_b = (const float*)d_in[5];
  float* out = (float*)d_out;

  // Workspace layout (bf16): Q | K | V^T | attn_out, each B*H*T*64 = 8388608
  bf16_t* ws = (bf16_t*)d_ws;
  const size_t nq = (size_t)kB * kH * kT * kDh;
  bf16_t* qws  = ws;
  bf16_t* kws  = ws + nq;
  bf16_t* vtws = ws + 2 * nq;
  bf16_t* abf  = ws + 3 * nq;

  mha_qkv_kernel<<<dim3(kN3 / 128, (kB * kT) / 128), 256, 0, stream>>>(
      x, qkv_w, qkv_b, qws, kws, vtws);
  mha_attn_kernel<<<dim3(kB * kH, kT / 128), 256, 0, stream>>>(
      qws, kws, vtws, mask, abf);
  mha_proj_kernel<<<dim3(kDim / 128, (kB * kT) / 128), 256, 0, stream>>>(
      abf, proj_w, proj_b, out);
}
